// GAT_RPE_L2L_Encoder_32581621907889
// MI455X (gfx1250) — matrix operand order repaired and checked by reading, not compiled, not run on hardware
//
#include <hip/hip_runtime.h>

#define NN 10000      // nodes
#define NE 160000     // edges
#define DD 128        // model dim
#define LL 3          // layers
#define HH 8          // heads

typedef __attribute__((ext_vector_type(16))) _Float16 v16h;
typedef __attribute__((ext_vector_type(8)))  float    v8f;

union AF { v16h v; _Float16 h[16]; };
union CF { v8f  v; float     f[8];  };

// ---------------- WMMA fragment helpers (f32->f16 path, 16x16x32) ----------------
// A-fragment (16x32 f16, MxK): lane&15 = row M; low lanes hold K=[k0..k0+7],[k0+16..k0+23],
// high lanes hold K=[k0+8..k0+15],[k0+24..k0+31]  (ISA 7.12.2 layout).
__device__ __forceinline__ v16h load_a_frag(const _Float16* As, int ldk, int k0, int lane) {
  AF a;
  const int row = lane & 15;
  const int off = (lane < 16) ? 0 : 8;
  const _Float16* p = As + row * ldk + k0 + off;
#pragma unroll
  for (int i = 0; i < 8; ++i) { a.h[i] = p[i]; a.h[8 + i] = p[16 + i]; }
  return a.v;
}

// B-fragment: weights pre-swizzled so each lane reads 16 contiguous halves (32B).
__device__ __forceinline__ v16h load_b_frag(const _Float16* Wz, int NT, int kt, int nt, int lane) {
  return *(const v16h*)(Wz + (((size_t)(kt * NT + nt) * 32 + lane) << 4));
}

__device__ __forceinline__ void wave_gemm(CF& acc, const _Float16* As, int ldk, int Ktot,
                                          const _Float16* Wz, int NT, int nt, int lane) {
  for (int kt = 0; kt < (Ktot >> 5); ++kt) {
    v16h a = load_a_frag(As, ldk, kt << 5, lane);
    v16h b = load_b_frag(Wz, NT, kt, nt, lane);
    acc.v = __builtin_amdgcn_wmma_f32_16x16x32_f16(false, a, false, b, (short)0, acc.v,
                                                   false, false);
  }
}

// C/D layout: lane<16 -> col nt*16+lane, rows 0..7 ; lane>=16 -> col nt*16+lane-16, rows 8..15
__device__ __forceinline__ void store_c(float* Cs, int ldc, int nt, int lane, const CF& acc) {
  const int col = nt * 16 + (lane & 15);
  const int rb  = (lane < 16) ? 0 : 8;
#pragma unroll
  for (int r = 0; r < 8; ++r) Cs[(rb + r) * ldc + col] = acc.f[r];
}

// per-row mean / rstd over DD columns for a 16-row LDS tile
__device__ __forceinline__ void block_rowstats(const float* Cs, int ldc, float* mean, float* rstd) {
  const int t = threadIdx.x;
  if (t < 16) {
    float s = 0.f, s2 = 0.f;
    for (int c = 0; c < DD; ++c) { float v = Cs[t * ldc + c]; s += v; s2 += v * v; }
    float m = s / DD;
    mean[t] = m;
    rstd[t] = rsqrtf(s2 / DD - m * m + 1e-5f);
  }
  __syncthreads();
}

__device__ __forceinline__ void atomicMaxF(float* addr, float val) {
  if (val >= 0.f) atomicMax((int*)addr, __float_as_int(val));
  else            atomicMin((unsigned int*)addr, __float_as_uint(val));
}

// ---------------- weight swizzle: fp32 [K,N] row-major -> f16 B-fragment order ----------------
__global__ void k_swizzle(const float* __restrict__ src, _Float16* __restrict__ dst,
                          int K, int Ncols) {
  int idx = blockIdx.x * blockDim.x + threadIdx.x;
  if (idx >= K * Ncols) return;
  int i    = idx & 15;
  int lane = (idx >> 4) & 31;
  int blk  = idx >> 9;              // kt*NT + nt
  int NT   = Ncols >> 4;
  int kt   = blk / NT, nt = blk - kt * NT;
  int k    = kt * 32 + ((i >> 3) << 4) + ((lane >> 4) << 3) + (i & 7);
  int n    = nt * 16 + (lane & 15);
  dst[idx] = (_Float16)src[(size_t)k * Ncols + n];
}

// ---------------- input projections: relu(ln(feats @ W + b)), K small -> VALU ----------------
__global__ __launch_bounds__(128) void k_proj(const float* __restrict__ feats, int Kin,
                                              const float* __restrict__ W, const float* __restrict__ b,
                                              const float* __restrict__ g, const float* __restrict__ bt,
                                              float* __restrict__ out) {
  __shared__ float red[128];
  const int row = blockIdx.x, c = threadIdx.x;
  const float* f = feats + (size_t)row * Kin;
  float s = b[c];
  for (int k = 0; k < Kin; ++k) s += f[k] * W[k * DD + c];
  red[c] = s; __syncthreads();
  for (int o = 64; o > 0; o >>= 1) { if (c < o) red[c] += red[c + o]; __syncthreads(); }
  float m = red[0] / DD; __syncthreads();
  red[c] = (s - m) * (s - m); __syncthreads();
  for (int o = 64; o > 0; o >>= 1) { if (c < o) red[c] += red[c + o]; __syncthreads(); }
  float rs = rsqrtf(red[0] / DD + 1e-5f);
  out[(size_t)row * DD + c] = fmaxf((s - m) * rs * g[c] + bt[c], 0.f);
}

// ---------------- mem = relu(ln([x_i|x_j|e] @ W + b)) : K=384 WMMA GEMM ----------------
__global__ __launch_bounds__(256) void k_mem(const float* __restrict__ x, const float* __restrict__ ef,
                                             const int* __restrict__ eidx, const _Float16* __restrict__ Wz,
                                             const float* __restrict__ bias, const float* __restrict__ g,
                                             const float* __restrict__ bt, _Float16* __restrict__ memb) {
  __shared__ _Float16 As[16 * 384];
  __shared__ float Cs[16 * 128];
  __shared__ float mean[16], rstd[16];
  const int e0 = blockIdx.x * 16;
  const int t = threadIdx.x, lane = t & 31, wave = t >> 5;
  for (int idx = t; idx < 16 * 384; idx += 256) {
    int r = idx / 384, c = idx - r * 384, eI = e0 + r;
    float v;
    if (c < 128)      v = x[(size_t)eidx[NE + eI] * DD + c];          // x_i = x[dst]
    else if (c < 256) v = x[(size_t)eidx[eI] * DD + (c - 128)];       // x_j = x[src]
    else              v = ef[(size_t)eI * DD + (c - 256)];
    As[idx] = (_Float16)v;
  }
  __syncthreads();
  CF acc;
  float bcol = bias[wave * 16 + (lane & 15)];
#pragma unroll
  for (int r = 0; r < 8; ++r) acc.f[r] = bcol;
  wave_gemm(acc, As, 384, 384, Wz, 8, wave, lane);
  store_c(Cs, 128, wave, lane, acc);
  __syncthreads();
  block_rowstats(Cs, 128, mean, rstd);
  for (int idx = t; idx < 16 * 128; idx += 256) {
    int r = idx >> 7, c = idx & 127;
    float v = fmaxf((Cs[idx] - mean[r]) * rstd[r] * g[c] + bt[c], 0.f);
    memb[(size_t)e0 * DD + idx] = (_Float16)v;
  }
}

// ---------------- delta = relu(ln(mem@euW+b)); e = ln(e+delta) ----------------
__global__ __launch_bounds__(256) void k_eupd(const _Float16* __restrict__ memb, float* __restrict__ ef,
                                              const _Float16* __restrict__ Wz, const float* __restrict__ bias,
                                              const float* __restrict__ eug, const float* __restrict__ eubt,
                                              const float* __restrict__ eng, const float* __restrict__ enbt) {
  __shared__ _Float16 As[16 * 128];
  __shared__ float Cs[16 * 128];
  __shared__ float mean[16], rstd[16];
  const int e0 = blockIdx.x * 16;
  const int t = threadIdx.x, lane = t & 31, wave = t >> 5;
  for (int idx = t; idx < 16 * 128; idx += 256) As[idx] = memb[(size_t)e0 * DD + idx];
  __syncthreads();
  CF acc;
  float bcol = bias[wave * 16 + (lane & 15)];
#pragma unroll
  for (int r = 0; r < 8; ++r) acc.f[r] = bcol;
  wave_gemm(acc, As, 128, 128, Wz, 8, wave, lane);
  store_c(Cs, 128, wave, lane, acc);
  __syncthreads();
  block_rowstats(Cs, 128, mean, rstd);
  for (int idx = t; idx < 16 * 128; idx += 256) {
    int r = idx >> 7, c = idx & 127;
    float d = fmaxf((Cs[idx] - mean[r]) * rstd[r] * eug[c] + eubt[c], 0.f);
    Cs[idx] = ef[(size_t)e0 * DD + idx] + d;   // residual
  }
  __syncthreads();
  block_rowstats(Cs, 128, mean, rstd);
  for (int idx = t; idx < 16 * 128; idx += 256) {
    int r = idx >> 7, c = idx & 127;
    ef[(size_t)e0 * DD + idx] = (Cs[idx] - mean[r]) * rstd[r] * eng[c] + enbt[c];
  }
}

// ---------------- q/k/v GEMMs + per-head logits ----------------
__global__ __launch_bounds__(256) void k_qkv(const float* __restrict__ x, const _Float16* __restrict__ memb,
                                             const int* __restrict__ eidx,
                                             const _Float16* __restrict__ Wqz, const _Float16* __restrict__ Wkz,
                                             const _Float16* __restrict__ Wvz,
                                             _Float16* __restrict__ vb, float* __restrict__ logits) {
  __shared__ _Float16 Ax[16 * 128];
  __shared__ _Float16 Am[16 * 128];
  __shared__ float Qs[16 * 128];
  __shared__ float Ks[16 * 128];
  const int e0 = blockIdx.x * 16;
  const int t = threadIdx.x, lane = t & 31, wave = t >> 5;
  for (int idx = t; idx < 16 * 128; idx += 256) {
    int r = idx >> 7, c = idx & 127;
    Ax[idx] = (_Float16)x[(size_t)eidx[NE + e0 + r] * DD + c];
    Am[idx] = memb[(size_t)e0 * DD + idx];
  }
  __syncthreads();
  CF q{}, k{}, cv{};
  wave_gemm(q,  Ax, 128, 128, Wqz, 8, wave, lane);
  wave_gemm(k,  Am, 128, 128, Wkz, 8, wave, lane);
  wave_gemm(cv, Am, 128, 128, Wvz, 8, wave, lane);
  store_c(Qs, 128, wave, lane, q);
  store_c(Ks, 128, wave, lane, k);
  {
    const int col = wave * 16 + (lane & 15);
    const int rb  = (lane < 16) ? 0 : 8;
#pragma unroll
    for (int r = 0; r < 8; ++r) vb[(size_t)(e0 + rb + r) * DD + col] = (_Float16)cv.f[r];
  }
  __syncthreads();
  if (t < 128) {
    int r = t >> 3, h = t & 7;
    float s = 0.f;
    for (int d = 0; d < 16; ++d) s += Qs[r * 128 + h * 16 + d] * Ks[r * 128 + h * 16 + d];
    logits[(size_t)(e0 + r) * HH + h] = s * 0.25f;   // 1/sqrt(16)
  }
}

// ---------------- segment softmax helpers ----------------
__global__ void k_init(float* mx, float* sm, float* agg) {
  int i = blockIdx.x * blockDim.x + threadIdx.x;
  if (i < NN * HH) { mx[i] = __int_as_float(0xff800000); sm[i] = 0.f; }
  if (i < NN * DD) agg[i] = 0.f;
}

__global__ void k_segmax(const float* __restrict__ logits, const int* __restrict__ eidx,
                         float* __restrict__ mx) {
  int i = blockIdx.x * blockDim.x + threadIdx.x;
  if (i >= NE * HH) return;
  int e = i >> 3, h = i & 7;
  atomicMaxF(&mx[eidx[NE + e] * HH + h], logits[i]);
}

__global__ void k_segexp(float* __restrict__ logits, const int* __restrict__ eidx,
                         const float* __restrict__ mx, float* __restrict__ sm) {
  int i = blockIdx.x * blockDim.x + threadIdx.x;
  if (i >= NE * HH) return;
  int e = i >> 3, h = i & 7;
  float ex = __expf(logits[i] - mx[eidx[NE + e] * HH + h]);
  logits[i] = ex;
  atomicAdd(&sm[eidx[NE + e] * HH + h], ex);
}

// ---------------- out_e = (attn*v) @ Wo ; scatter-add into agg ----------------
__global__ __launch_bounds__(256) void k_out(const _Float16* __restrict__ vb, const float* __restrict__ exb,
                                             const float* __restrict__ sm, const int* __restrict__ eidx,
                                             const _Float16* __restrict__ Woz, float* __restrict__ agg) {
  __shared__ _Float16 As[16 * 128];
  __shared__ int dsts[16];
  const int e0 = blockIdx.x * 16;
  const int t = threadIdx.x, lane = t & 31, wave = t >> 5;
  if (t < 16) dsts[t] = eidx[NE + e0 + t];
  __syncthreads();
  for (int idx = t; idx < 16 * 128; idx += 256) {
    int r = idx >> 7, c = idx & 127, h = c >> 4;
    float attn = exb[(size_t)(e0 + r) * HH + h] / (sm[dsts[r] * HH + h] + 1e-16f);
    As[idx] = (_Float16)(attn * (float)vb[(size_t)e0 * DD + idx]);
  }
  __syncthreads();
  CF acc{};
  wave_gemm(acc, As, 128, 128, Woz, 8, wave, lane);
  const int col = wave * 16 + (lane & 15);
  const int rb  = (lane < 16) ? 0 : 8;
#pragma unroll
  for (int r = 0; r < 8; ++r) atomicAdd(&agg[(size_t)dsts[rb + r] * DD + col], acc.f[r]);
}

// ---------------- node: x=ln(x+agg); h=relu(x@W1+b1)@W2+b2; x=ln(x+h) ----------------
__global__ __launch_bounds__(256) void k_node(float* __restrict__ x, const float* __restrict__ agg,
                                              const _Float16* __restrict__ W1z, const float* __restrict__ b1,
                                              const _Float16* __restrict__ W2z, const float* __restrict__ b2,
                                              const float* __restrict__ n1g, const float* __restrict__ n1bt,
                                              const float* __restrict__ n2g, const float* __restrict__ n2bt) {
  __shared__ float Xs[16 * 128];
  __shared__ _Float16 Ah[16 * 128];
  __shared__ _Float16 Hs[16 * 256];
  __shared__ float mean[16], rstd[16];
  const int n0 = blockIdx.x * 16;
  const int t = threadIdx.x, lane = t & 31, wave = t >> 5;
  for (int idx = t; idx < 16 * 128; idx += 256)
    Xs[idx] = x[(size_t)n0 * DD + idx] + agg[(size_t)n0 * DD + idx];
  __syncthreads();
  block_rowstats(Xs, 128, mean, rstd);
  for (int idx = t; idx < 16 * 128; idx += 256) {
    int r = idx >> 7, c = idx & 127;
    float v = (Xs[idx] - mean[r]) * rstd[r] * n1g[c] + n1bt[c];
    Xs[idx] = v;
    Ah[idx] = (_Float16)v;
  }
  __syncthreads();
  // FFN GEMM1: 16x256, each wave handles 2 column tiles
  for (int nt = wave; nt < 16; nt += 8) {
    CF acc;
    float bcol = b1[nt * 16 + (lane & 15)];
#pragma unroll
    for (int r = 0; r < 8; ++r) acc.f[r] = bcol;
    wave_gemm(acc, Ah, 128, 128, W1z, 16, nt, lane);
    const int col = nt * 16 + (lane & 15);
    const int rb  = (lane < 16) ? 0 : 8;
#pragma unroll
    for (int r = 0; r < 8; ++r) Hs[(rb + r) * 256 + col] = (_Float16)fmaxf(acc.f[r], 0.f);
  }
  __syncthreads();
  // FFN GEMM2: K=256 -> 16x128, residual added into Xs
  CF acc2;
  float bcol2 = b2[wave * 16 + (lane & 15)];
#pragma unroll
  for (int r = 0; r < 8; ++r) acc2.f[r] = bcol2;
  wave_gemm(acc2, Hs, 256, 256, W2z, 8, wave, lane);
  {
    const int col = wave * 16 + (lane & 15);
    const int rb  = (lane < 16) ? 0 : 8;
#pragma unroll
    for (int r = 0; r < 8; ++r) Xs[(rb + r) * 128 + col] += acc2.f[r];
  }
  __syncthreads();
  block_rowstats(Xs, 128, mean, rstd);
  for (int idx = t; idx < 16 * 128; idx += 256) {
    int r = idx >> 7, c = idx & 127;
    x[(size_t)n0 * DD + idx] = (Xs[idx] - mean[r]) * rstd[r] * n2g[c] + n2bt[c];
  }
}

// =====================================================================================
extern "C" void kernel_launch(void* const* d_in, const int* in_sizes, int n_in,
                              void* d_out, int out_size, void* d_ws, size_t ws_size,
                              hipStream_t stream) {
  (void)in_sizes; (void)n_in; (void)out_size; (void)ws_size;
  const float* lane_feats = (const float*)d_in[0];
  const float* edge_attrs = (const float*)d_in[1];
  const int*   eidx       = (const int*)d_in[2];
  const float* node_W = (const float*)d_in[3];
  const float* node_b = (const float*)d_in[4];
  const float* node_g = (const float*)d_in[5];
  const float* node_bt= (const float*)d_in[6];
  const float* rpe_W  = (const float*)d_in[7];
  const float* rpe_b  = (const float*)d_in[8];
  const float* rpe_g  = (const float*)d_in[9];
  const float* rpe_bt = (const float*)d_in[10];
  const float* mem_W  = (const float*)d_in[11];
  const float* mem_b  = (const float*)d_in[12];
  const float* mem_g  = (const float*)d_in[13];
  const float* mem_bt = (const float*)d_in[14];
  const float* Wq     = (const float*)d_in[15];
  const float* Wk     = (const float*)d_in[16];
  const float* Wv     = (const float*)d_in[17];
  const float* Wo     = (const float*)d_in[18];
  const float* eu_W   = (const float*)d_in[19];
  const float* eu_b   = (const float*)d_in[20];
  const float* eu_g   = (const float*)d_in[21];
  const float* eu_bt  = (const float*)d_in[22];
  const float* en_g   = (const float*)d_in[23];
  const float* en_bt  = (const float*)d_in[24];
  const float* ffn_W1 = (const float*)d_in[25];
  const float* ffn_b1 = (const float*)d_in[26];
  const float* ffn_W2 = (const float*)d_in[27];
  const float* ffn_b2 = (const float*)d_in[28];
  const float* n1_g   = (const float*)d_in[29];
  const float* n1_bt  = (const float*)d_in[30];
  const float* n2_g   = (const float*)d_in[31];
  const float* n2_bt  = (const float*)d_in[32];

  float* x = (float*)d_out;                       // [N,128] node features live in d_out

  // workspace layout
  size_t off = 0;
  char* ws = (char*)d_ws;
  auto take = [&](size_t bytes) { size_t o = off; off = (off + bytes + 255) & ~(size_t)255; return o; };
  float*    ef     = (float*)   (ws + take((size_t)NE * DD * 4));
  _Float16* memb   = (_Float16*)(ws + take((size_t)NE * DD * 2));
  _Float16* vb     = (_Float16*)(ws + take((size_t)NE * DD * 2));
  float*    logits = (float*)   (ws + take((size_t)NE * HH * 4));
  float*    mx     = (float*)   (ws + take((size_t)NN * HH * 4));
  float*    sm     = (float*)   (ws + take((size_t)NN * HH * 4));
  float*    agg    = (float*)   (ws + take((size_t)NN * DD * 4));
  _Float16* memWz  = (_Float16*)(ws + take((size_t)LL * 384 * DD * 2));
  _Float16* euz    = (_Float16*)(ws + take((size_t)LL * DD * DD * 2));
  _Float16* qz     = (_Float16*)(ws + take((size_t)LL * DD * DD * 2));
  _Float16* kz     = (_Float16*)(ws + take((size_t)LL * DD * DD * 2));
  _Float16* vz     = (_Float16*)(ws + take((size_t)LL * DD * DD * 2));
  _Float16* oz     = (_Float16*)(ws + take((size_t)LL * DD * DD * 2));
  _Float16* w1z    = (_Float16*)(ws + take((size_t)LL * DD * 256 * 2));
  _Float16* w2z    = (_Float16*)(ws + take((size_t)LL * 256 * DD * 2));

  auto swz = [&](const float* src, _Float16* dst, int K, int Ncols) {
    int total = K * Ncols;
    k_swizzle<<<(total + 255) / 256, 256, 0, stream>>>(src, dst, K, Ncols);
  };
  for (int l = 0; l < LL; ++l) {
    swz(mem_W  + (size_t)l * 384 * DD, memWz + (size_t)l * 384 * DD, 384, DD);
    swz(eu_W   + (size_t)l * DD * DD,  euz   + (size_t)l * DD * DD,  DD, DD);
    swz(Wq     + (size_t)l * DD * DD,  qz    + (size_t)l * DD * DD,  DD, DD);
    swz(Wk     + (size_t)l * DD * DD,  kz    + (size_t)l * DD * DD,  DD, DD);
    swz(Wv     + (size_t)l * DD * DD,  vz    + (size_t)l * DD * DD,  DD, DD);
    swz(Wo     + (size_t)l * DD * DD,  oz    + (size_t)l * DD * DD,  DD, DD);
    swz(ffn_W1 + (size_t)l * DD * 256, w1z   + (size_t)l * DD * 256, DD, 256);
    swz(ffn_W2 + (size_t)l * 256 * DD, w2z   + (size_t)l * 256 * DD, 256, DD);
  }

  // input projections
  k_proj<<<NN, 128, 0, stream>>>(lane_feats, 10, node_W, node_b, node_g, node_bt, x);
  k_proj<<<NE, 128, 0, stream>>>(edge_attrs, 9, rpe_W, rpe_b, rpe_g, rpe_bt, ef);

  const int ET = NE / 16, NT = NN / 16;
  for (int l = 0; l < LL; ++l) {
    k_mem <<<ET, 256, 0, stream>>>(x, ef, eidx, memWz + (size_t)l * 384 * DD,
                                   mem_b + (size_t)l * DD, mem_g + (size_t)l * DD,
                                   mem_bt + (size_t)l * DD, memb);
    k_eupd<<<ET, 256, 0, stream>>>(memb, ef, euz + (size_t)l * DD * DD,
                                   eu_b + (size_t)l * DD, eu_g + (size_t)l * DD,
                                   eu_bt + (size_t)l * DD,
                                   en_g + (size_t)l * DD, en_bt + (size_t)l * DD);
    k_qkv <<<ET, 256, 0, stream>>>(x, memb, eidx,
                                   qz + (size_t)l * DD * DD, kz + (size_t)l * DD * DD,
                                   vz + (size_t)l * DD * DD, vb, logits);
    k_init<<<(NN * DD + 255) / 256, 256, 0, stream>>>(mx, sm, agg);
    k_segmax<<<(NE * HH + 255) / 256, 256, 0, stream>>>(logits, eidx, mx);
    k_segexp<<<(NE * HH + 255) / 256, 256, 0, stream>>>(logits, eidx, mx, sm);
    k_out <<<ET, 256, 0, stream>>>(vb, logits, sm, eidx, oz + (size_t)l * DD * DD, agg);
    k_node<<<NT, 256, 0, stream>>>(x, agg,
                                   w1z + (size_t)l * DD * 256, ffn_b1 + (size_t)l * 256,
                                   w2z + (size_t)l * 256 * DD, ffn_b2 + (size_t)l * DD,
                                   n1_g + (size_t)l * DD, n1_bt + (size_t)l * DD,
                                   n2_g + (size_t)l * DD, n2_bt + (size_t)l * DD);
  }
}
